// Decoder_34205119545704
// MI455X (gfx1250) — matrix operand order repaired and checked by reading, not compiled, not run on hardware
//
#include <hip/hip_runtime.h>
#include <hip/hip_bf16.h>
#include <math.h>

// ---------------------------------------------------------------------------
// Decoder: fc1+relu -> 3x GRU(H=501, T=277) -> fc2 -> (B,O,T)
// MI455X / gfx1250, wave32, WMMA bf16 (f32 accumulate).
//  * batch-partitioned persistent GRU kernel: 16 WGs x 16 batch rows, all 277
//    timesteps inside the kernel -> zero cross-WG sync; weights L2-resident.
//  * gate dim padded to 96 N-tiles = 16 waves x exactly 6 tiles (static
//    tiling: no divergent EXEC around WMMA, no spilled accumulators).
//  * weights pre-tiled kt-major so each wave's 6 B-fragments per k-step are
//    one contiguous 6KB run (clause of global_load_b128 w/ imm offsets).
// Workspace requirement: ~82 MB.
// ---------------------------------------------------------------------------

typedef __bf16 v16bf __attribute__((ext_vector_type(16)));
typedef float  v8f   __attribute__((ext_vector_type(8)));

#define DD   56     // latent size
#define Hh   501    // hidden
#define HP   512    // hidden padded (WMMA K padding)
#define G3   1503   // 3*H
#define TT   277    // timesteps
#define BB   256    // batch
#define OO   76     // fc2 out
#define BM   16     // batch rows per workgroup (WMMA M)
#define NTP  96     // gate N-tiles padded (1536/16) = 16 waves * 6
#define GW   1536   // padded gate width (NTP*16)
#define KT   16     // K-tiles (512/32)
#define OTP  5      // fc2 O-tiles (80/16)
#define WAVES   16
#define THREADS 512
#define TPW  6      // tiles per wave (static!)

// contiguous 16-bf16 fragment (tiled weights: 32B per lane)
__device__ __forceinline__ v16bf load_frag_contig(const __bf16* p) {
    union { v16bf v; uint4 u[2]; } r;
    r.u[0] = *(const uint4*)(p);
    r.u[1] = *(const uint4*)(p + 8);
    return r.v;
}
// A-fragment: two 8-bf16 chunks at K = base and K = base+16 (ISA 7.12.2)
__device__ __forceinline__ v16bf load_frag_split(const __bf16* p0, const __bf16* p1) {
    union { v16bf v; uint4 u[2]; } r;
    r.u[0] = *(const uint4*)(p0);
    r.u[1] = *(const uint4*)(p1);
    return r.v;
}

__device__ __forceinline__ float sigmoidf_(float x) {
    return 1.0f / (1.0f + __expf(-x));
}

// ---------------------------------------------------------------------------
// Weight tiling: W (N x K, f32 row-major) -> bf16 B-fragments, kt-major:
// out[((kt*ntp + nt)*32 + lane)*16 + e] =
//        W[nt*16 + lane%16][kt*32 + (lane/16)*8 + (e%8) + (e/8)*16]   (zero padded)
// ---------------------------------------------------------------------------
__global__ void tile_weights(const float* __restrict__ W, __bf16* __restrict__ out,
                             int N, int K, int ntp, int ktc) {
    int id = blockIdx.x * blockDim.x + threadIdx.x;
    int total = ntp * ktc * 512;
    if (id >= total) return;
    int e    = id & 15;
    int lane = (id >> 4) & 31;
    int nt   = (id >> 9) % ntp;
    int kt   = (id >> 9) / ntp;
    int n  = nt * 16 + (lane & 15);
    int kk = kt * 32 + (lane >> 4) * 8 + (e & 7) + (e >> 3) * 16;
    float v = (n < N && kk < K) ? W[(size_t)n * K + kk] : 0.0f;
    out[id] = (__bf16)v;
}

// ---------------------------------------------------------------------------
// fc1 + relu  (tiny: 256x56x56)
// ---------------------------------------------------------------------------
__global__ void fc1_relu(const float* __restrict__ x, const float* __restrict__ w,
                         const float* __restrict__ b, float* __restrict__ hin) {
    int id = blockIdx.x * blockDim.x + threadIdx.x;
    if (id >= BB * DD) return;
    int o = id % DD, bi = id / DD;
    float s = b[o];
    for (int k = 0; k < DD; ++k) s += x[bi * DD + k] * w[o * DD + k];
    hin[id] = s > 0.0f ? s : 0.0f;
}

// g0 = hin @ w_ih0^T + b_ih0  (constant over t; 256x1503x56)
__global__ void g0_kernel(const float* __restrict__ hin, const float* __restrict__ w,
                          const float* __restrict__ b, float* __restrict__ g0) {
    int id = blockIdx.x * blockDim.x + threadIdx.x;
    if (id >= BB * G3) return;
    int g = id % G3, bi = id / G3;
    float s = b[g];
    for (int k = 0; k < DD; ++k) s += hin[bi * DD + k] * w[g * DD + k];
    g0[id] = s;
}

// ---------------------------------------------------------------------------
// Persistent GRU layer: grid = 16 WGs (16 batch rows each), 512 thr = 16 waves.
// Wave w owns N-tiles [w*6, w*6+6).  Per step:
//   phase 1a: gh = h@Whh^T  (6 static WMMA accumulators)
//   phase 1b: gx = x_t@Wih^T (layers 1,2)
//   phase 2 : gate fusion in LDS, h updated, output -> ys[t]
// ys (T,B,512) bf16 is read and overwritten in place (batch rows exclusive
// per WG, read-before-write within the step); pad cols [501,512) kept zero.
// ---------------------------------------------------------------------------
__global__ __launch_bounds__(THREADS) void gru_layer(
    const __bf16* __restrict__ whh_t,   // tiled [KT][NTP][32][16]
    const __bf16* __restrict__ wih_t,   // tiled (layers 1,2) or unused
    const float*  __restrict__ g0,      // (B,3H) layer 0 input gates, or unused
    const float*  __restrict__ b_ih,    // (3H)   layers 1,2
    const float*  __restrict__ b_hh,    // (3H)
    __bf16*       __restrict__ ys,      // (T,B,HP) bf16, in/out
    int has_input)
{
    extern __shared__ char smem[];
    float*  gh = (float*)smem;               // [BM][GW]  hidden-GEMM gates
    float*  gx = gh + BM * GW;               // [BM][GW]  input-GEMM gates
    float*  hf = gx + BM * GW;               // [BM][HP]  h state, f32
    __bf16* hb = (__bf16*)(hf + BM * HP);    // [BM][HP]  h state, bf16 mirror

    const int tid  = threadIdx.x;
    const int lane = tid & 31;
    const int wave = tid >> 5;
    const int b0   = blockIdx.x * BM;
    const int nt0  = wave * TPW;             // first of this wave's 6 N-tiles

    // init h = 0 (incl. K padding)
    for (int i = tid; i < BM * HP; i += THREADS) { hf[i] = 0.0f; hb[i] = (__bf16)0.0f; }
    __syncthreads();

    const int am    = lane & 15;        // A-fragment row (batch-slice row)
    const int abase = (lane >> 4) * 8;  // A-fragment K base
    const int mrow  = (lane >> 4) * 8;  // C-fragment first M row
    const int c0    = nt0 * 16 + (lane & 15);

    // per-lane base into tiled weights: tile (kt,nt0+i) at + kt*NTP*512 + i*512
    const __bf16* whh_lane = whh_t + ((size_t)nt0 * 32 + lane) * 16;
    const __bf16* wih_lane = wih_t + ((size_t)nt0 * 32 + lane) * 16;

    const v8f vzero = {};

    for (int t = 0; t < TT; ++t) {
        // ---------- phase 1a: hidden GEMM  gh = h @ Whh^T ----------
        {
            v8f acc[TPW];
            #pragma unroll
            for (int i = 0; i < TPW; ++i) acc[i] = vzero;
            for (int kt = 0; kt < KT; ++kt) {
                const __bf16* hr = hb + am * HP + kt * 32;
                v16bf a = load_frag_split(hr + abase, hr + abase + 16);
                const __bf16* bp = whh_lane + (size_t)kt * NTP * 512;
                v16bf bf[TPW];
                #pragma unroll
                for (int i = 0; i < TPW; ++i) bf[i] = load_frag_contig(bp + i * 512);
                #pragma unroll
                for (int i = 0; i < TPW; ++i)
                    acc[i] = __builtin_amdgcn_wmma_f32_16x16x32_bf16(
                        false, a, false, bf[i], (short)0, acc[i], false, false);
            }
            #pragma unroll
            for (int i = 0; i < TPW; ++i) {
                #pragma unroll
                for (int r = 0; r < 8; ++r)
                    gh[(mrow + r) * GW + c0 + i * 16] = acc[i][r];
            }
        }

        // ---------- phase 1b: input GEMM  gx = x_t @ Wih^T (layers 1,2) ----------
        if (has_input) {
            v8f acc[TPW];
            #pragma unroll
            for (int i = 0; i < TPW; ++i) acc[i] = vzero;
            const __bf16* xrow = ys + ((size_t)t * BB + b0 + am) * HP;
            for (int kt = 0; kt < KT; ++kt) {
                v16bf a = load_frag_split(xrow + kt * 32 + abase,
                                          xrow + kt * 32 + abase + 16);
                const __bf16* bp = wih_lane + (size_t)kt * NTP * 512;
                v16bf bf[TPW];
                #pragma unroll
                for (int i = 0; i < TPW; ++i) bf[i] = load_frag_contig(bp + i * 512);
                #pragma unroll
                for (int i = 0; i < TPW; ++i)
                    acc[i] = __builtin_amdgcn_wmma_f32_16x16x32_bf16(
                        false, a, false, bf[i], (short)0, acc[i], false, false);
            }
            #pragma unroll
            for (int i = 0; i < TPW; ++i) {
                #pragma unroll
                for (int r = 0; r < 8; ++r)
                    gx[(mrow + r) * GW + c0 + i * 16] = acc[i][r];
            }
        }
        __syncthreads();

        // ---------- phase 2: gate fusion (PyTorch GRU semantics) ----------
        for (int idx = tid; idx < BM * HP; idx += THREADS) {
            int m = idx >> 9;         // / HP
            int j = idx & (HP - 1);   // % HP
            __bf16* yout = ys + ((size_t)t * BB + b0 + m) * HP + j;
            if (j < Hh) {
                float ghr = gh[m * GW + j];
                float ghz = gh[m * GW + Hh + j];
                float ghn = gh[m * GW + 2 * Hh + j];
                float gxr, gxz, gxn;
                if (has_input) {
                    gxr = gx[m * GW + j]          + b_ih[j];
                    gxz = gx[m * GW + Hh + j]     + b_ih[Hh + j];
                    gxn = gx[m * GW + 2 * Hh + j] + b_ih[2 * Hh + j];
                } else {
                    const float* gr = g0 + (size_t)(b0 + m) * G3;
                    gxr = gr[j]; gxz = gr[Hh + j]; gxn = gr[2 * Hh + j];
                }
                float rr = sigmoidf_(gxr + ghr + b_hh[j]);
                float zz = sigmoidf_(gxz + ghz + b_hh[Hh + j]);
                float nn = tanhf(gxn + rr * (ghn + b_hh[2 * Hh + j]));
                float hnew = (1.0f - zz) * nn + zz * hf[idx];
                hf[idx] = hnew;
                hb[idx] = (__bf16)hnew;
                *yout   = (__bf16)hnew;
            } else {
                *yout = (__bf16)0.0f;   // keep K padding zero for next layer
            }
        }
        __syncthreads();
    }
}

// ---------------------------------------------------------------------------
// fc2: out[b][o][t] = ys2[t][b][:] . fc2_w[o][:] + b[o]   (WMMA, 1 tile/wave)
// ---------------------------------------------------------------------------
__global__ __launch_bounds__(256) void fc2_kernel(
    const __bf16* __restrict__ ys, const __bf16* __restrict__ w_t,
    const float* __restrict__ bias, float* __restrict__ out)
{
    int wid  = blockIdx.x * 8 + (threadIdx.x >> 5);   // grid sized exactly
    int lane = threadIdx.x & 31;
    int ot = wid % OTP;
    int bt = (wid / OTP) % (BB / 16);
    int t  = wid / (OTP * (BB / 16));

    const int am = lane & 15, abase = (lane >> 4) * 8;
    const __bf16* xrow = ys + ((size_t)t * BB + bt * 16 + am) * HP;
    const __bf16* wlane = w_t + ((size_t)ot * 32 + lane) * 16;
    v8f acc = {};
    for (int kt = 0; kt < KT; ++kt) {
        v16bf a = load_frag_split(xrow + kt * 32 + abase,
                                  xrow + kt * 32 + abase + 16);
        v16bf b = load_frag_contig(wlane + (size_t)kt * OTP * 512);
        acc = __builtin_amdgcn_wmma_f32_16x16x32_bf16(
            false, a, false, b, (short)0, acc, false, false);
    }
    int o = ot * 16 + (lane & 15);
    if (o < OO) {
        int mrow = (lane >> 4) * 8;
        float bo = bias[o];
        #pragma unroll
        for (int r = 0; r < 8; ++r) {
            int bidx = bt * 16 + mrow + r;
            out[((size_t)bidx * OO + o) * TT + t] = acc[r] + bo;
        }
    }
}

// ---------------------------------------------------------------------------
extern "C" void kernel_launch(void* const* d_in, const int* in_sizes, int n_in,
                              void* d_out, int out_size, void* d_ws, size_t ws_size,
                              hipStream_t stream) {
    const float* x      = (const float*)d_in[0];
    const float* fc1_w  = (const float*)d_in[1];
    const float* fc1_b  = (const float*)d_in[2];
    const float* w_ih0  = (const float*)d_in[3];
    const float* w_hh0  = (const float*)d_in[4];
    const float* b_ih0  = (const float*)d_in[5];
    const float* b_hh0  = (const float*)d_in[6];
    const float* w_ih1  = (const float*)d_in[7];
    const float* w_hh1  = (const float*)d_in[8];
    const float* b_ih1  = (const float*)d_in[9];
    const float* b_hh1  = (const float*)d_in[10];
    const float* w_ih2  = (const float*)d_in[11];
    const float* w_hh2  = (const float*)d_in[12];
    const float* b_ih2  = (const float*)d_in[13];
    const float* b_hh2  = (const float*)d_in[14];
    const float* fc2_w  = (const float*)d_in[15];
    const float* fc2_b  = (const float*)d_in[16];
    float* out = (float*)d_out;

    // ---- carve workspace (needs ~82 MB) ----
    char* ws = (char*)d_ws;
    size_t off = 0;
    auto carve = [&](size_t bytes) -> char* {
        off = (off + 255) & ~(size_t)255;
        char* p = ws + off;
        off += bytes;
        return p;
    };
    const size_t gwsz = (size_t)KT * NTP * 512 * sizeof(__bf16);
    __bf16* ys      = (__bf16*)carve((size_t)TT * BB * HP * sizeof(__bf16));
    __bf16* whh_t0  = (__bf16*)carve(gwsz);
    __bf16* whh_t1  = (__bf16*)carve(gwsz);
    __bf16* whh_t2  = (__bf16*)carve(gwsz);
    __bf16* wih_t1  = (__bf16*)carve(gwsz);
    __bf16* wih_t2  = (__bf16*)carve(gwsz);
    __bf16* fc2w_t  = (__bf16*)carve((size_t)KT * OTP * 512 * sizeof(__bf16));
    float*  g0buf   = (float*)carve((size_t)BB * G3 * sizeof(float));
    float*  hin     = (float*)carve((size_t)BB * DD * sizeof(float));

    // ---- weight tiling (one-time per call; tiny) ----
    {
        int tot = KT * NTP * 512;
        int blk = (tot + 255) / 256;
        tile_weights<<<blk, 256, 0, stream>>>(w_hh0, whh_t0, G3, Hh, NTP, KT);
        tile_weights<<<blk, 256, 0, stream>>>(w_hh1, whh_t1, G3, Hh, NTP, KT);
        tile_weights<<<blk, 256, 0, stream>>>(w_hh2, whh_t2, G3, Hh, NTP, KT);
        tile_weights<<<blk, 256, 0, stream>>>(w_ih1, wih_t1, G3, Hh, NTP, KT);
        tile_weights<<<blk, 256, 0, stream>>>(w_ih2, wih_t2, G3, Hh, NTP, KT);
        int tot2 = KT * OTP * 512;
        tile_weights<<<(tot2 + 255) / 256, 256, 0, stream>>>(fc2_w, fc2w_t, OO, Hh, OTP, KT);
    }

    // ---- fc1 + relu, layer-0 input gates (constant over t) ----
    fc1_relu<<<(BB * DD + 255) / 256, 256, 0, stream>>>(x, fc1_w, fc1_b, hin);
    g0_kernel<<<(BB * G3 + 255) / 256, 256, 0, stream>>>(hin, w_ih0, b_ih0, g0buf);

    // ---- GRU layers (batch-partitioned, persistent over T) ----
    size_t smem = (size_t)(2 * BM * GW + BM * HP) * sizeof(float)
                + (size_t)BM * HP * sizeof(__bf16);   // 240 KB of 320 KB LDS
    gru_layer<<<BB / BM, THREADS, smem, stream>>>(whh_t0, whh_t0, g0buf,
                                                  b_hh0 /*unused*/, b_hh0, ys, 0);
    gru_layer<<<BB / BM, THREADS, smem, stream>>>(whh_t1, wih_t1, g0buf /*unused*/,
                                                  b_ih1, b_hh1, ys, 1);
    gru_layer<<<BB / BM, THREADS, smem, stream>>>(whh_t2, wih_t2, g0buf /*unused*/,
                                                  b_ih2, b_hh2, ys, 1);

    // ---- fc2 + transpose to (B, O, T) ----
    int fc2_blocks = (TT * (BB / 16) * OTP) / 8;      // 22160 tiles / 8 waves
    fc2_kernel<<<fc2_blocks, 256, 0, stream>>>(ys, fc2w_t, fc2_b, out);
}
